// Qwen3FlashMTPAttention_16131897164045
// MI455X (gfx1250) — compile-verified, matrix operand order checked by reading, not run on hardware
//
#include <hip/hip_runtime.h>
#include <hip/hip_bf16.h>

typedef unsigned short u16;
typedef __attribute__((ext_vector_type(16))) __bf16 v16bf;
typedef __attribute__((ext_vector_type(8)))  float  v8f;

// pointer types for the gfx1250 async LDS-DMA builtin (signature revealed by
// probe: (global int4*, local int4*, imm offset, imm cpol))
typedef int v4i __attribute__((vector_size(16)));
typedef __attribute__((address_space(1))) v4i* gv4i;
typedef __attribute__((address_space(3))) v4i* lv4i;

#if defined(__has_builtin) && __has_builtin(__builtin_amdgcn_s_wait_asynccnt)
#define WAIT_ASYNC(n) __builtin_amdgcn_s_wait_asynccnt(n)
#else
#define WAIT_ASYNC(n) asm volatile("s_wait_asynccnt %0" ::"i"(n) : "memory")
#endif

constexpr int Bc   = 2;
constexpr int Sq   = 2048;
constexpr int Hd   = 2048;
constexpr int CTX  = 1024;
constexpr int MROT = Sq - CTX;   // 1024 rotated positions
constexpr int NH   = 16;
constexpr int NKV  = 8;
constexpr int HD   = 128;
constexpr float SCALE = 0.08838834764831845f;  // 128^-0.5

// ---------------------------------------------------------------- helpers ---
static __device__ __forceinline__ u16 f2bf(float f) {
  unsigned u = __float_as_uint(f);
  u = (u + 0x7FFFu + ((u >> 16) & 1u)) >> 16;   // round-to-nearest-even
  return (u16)u;
}

// A/B fragment for V_WMMA_F32_16X16X32_BF16: per lane, 8 contiguous bf16 at
// base and 8 contiguous bf16 at base+16 elements (two 16B loads).
static __device__ __forceinline__ v16bf ld_frag(const u16* p0, const u16* p1) {
  uint4 t[2];
  t[0] = *(const uint4*)p0;
  t[1] = *(const uint4*)p1;
  v16bf r;
  __builtin_memcpy(&r, t, 32);
  return r;
}

static __device__ __forceinline__ v8f wmma_bf16(v16bf a, v16bf b, v8f c) {
  return __builtin_amdgcn_wmma_f32_16x16x32_bf16(
      /*neg_a=*/false, a, /*neg_b=*/false, b,
      /*c_mod=*/(short)0, c, /*reuse_a=*/false, /*reuse_b=*/false);
}

// 16B async copy global -> LDS (tracked by ASYNCcnt)
static __device__ __forceinline__ void async_cp16(const u16* src, u16* lds) {
  __builtin_amdgcn_global_load_async_to_lds_b128((gv4i)src, (lv4i)lds, 0, 0);
}

// ----------------------------------------------------- TDM probe (unlaunched)
#if defined(__has_builtin)
#if __has_builtin(__builtin_amdgcn_tensor_load_to_lds)
typedef __attribute__((ext_vector_type(4))) unsigned tdm_u32x4;
typedef __attribute__((ext_vector_type(4))) int      tdm_i32x4;
typedef __attribute__((ext_vector_type(8))) int      tdm_i32x8;
__global__ void cdna5_probe_tdm() {
  tdm_u32x4 g0 = 0;
  tdm_i32x8 g1 = 0;
  tdm_i32x4 g2 = 0, g3 = 0;
#if __clang_major__ >= 23
  tdm_i32x8 g4 = 0;
  __builtin_amdgcn_tensor_load_to_lds(g0, g1, g2, g3, g4, 0);
#else
  __builtin_amdgcn_tensor_load_to_lds(g0, g1, g2, g3, 0);
#endif
#if __has_builtin(__builtin_amdgcn_s_wait_tensorcnt)
  __builtin_amdgcn_s_wait_tensorcnt(0);
#else
  asm volatile("s_wait_tensorcnt 0x0" ::: "memory");
#endif
}
#endif
#endif  // __has_builtin

// ------------------------------------------------------------ cvt fp32->bf16
__global__ void cvt_bf16(const float* __restrict__ in, u16* __restrict__ out,
                         size_t n) {
  size_t gid = (size_t)blockIdx.x * blockDim.x + threadIdx.x;
  if (gid < n) out[gid] = f2bf(in[gid]);
}

// --------------------------------------------------- GEMM: C = A * B^T (f32)
// A: MxK bf16 row-major, Bw: NxK bf16 row-major, C: MxN f32. N,K compile-time
// so the epilogue uses immediate store offsets. 256 threads = 8 waves; wave
// tile 32x64 (2x4 WMMA); block tile 64x256; double-buffered k-loop.
template <int K>
static __device__ __forceinline__ void load_ab(const u16* ap0, const u16* bp0,
                                               int k0, v16bf (&a)[2],
                                               v16bf (&b)[4]) {
#pragma unroll
  for (int i = 0; i < 2; ++i) {
    const u16* p = ap0 + (size_t)i * 16 * K + k0;
    a[i] = ld_frag(p, p + 16);
    __builtin_prefetch(p + 128, 0, 3);  // ~4 k-steps ahead
  }
#pragma unroll
  for (int j = 0; j < 4; ++j) {
    const u16* p = bp0 + (size_t)j * 16 * K + k0;
    b[j] = ld_frag(p, p + 16);
    __builtin_prefetch(p + 128, 0, 3);
  }
}

static __device__ __forceinline__ void mma_tile(v8f (&acc)[2][4],
                                                const v16bf (&a)[2],
                                                const v16bf (&b)[4]) {
#pragma unroll
  for (int i = 0; i < 2; ++i)
#pragma unroll
    for (int j = 0; j < 4; ++j)
      acc[i][j] = wmma_bf16(a[i], b[j], acc[i][j]);
}

template <int N, int K>
__global__ __launch_bounds__(256) void gemm_bf16nt(
    const u16* __restrict__ A, const u16* __restrict__ Bw,
    float* __restrict__ C) {
  const int lane = threadIdx.x & 31;
  const int wid  = threadIdx.x >> 5;
  const int col  = lane & 15;
  const int hi   = lane >> 4;
  const int m0 = blockIdx.y * 64  + (wid & 1) * 32;
  const int n0 = blockIdx.x * 256 + (wid >> 1) * 64;

  v8f acc[2][4];
#pragma unroll
  for (int i = 0; i < 2; ++i)
#pragma unroll
    for (int j = 0; j < 4; ++j)
      acc[i][j] = v8f{0.f, 0.f, 0.f, 0.f, 0.f, 0.f, 0.f, 0.f};

  const u16* ap0 = A  + (size_t)(m0 + col) * K + hi * 8;
  const u16* bp0 = Bw + (size_t)(n0 + col) * K + hi * 8;

  v16bf a0[2], b0[4], a1[2], b1[4];
  load_ab<K>(ap0, bp0, 0, a0, b0);
#pragma unroll 1
  for (int k0 = 0; k0 < K; k0 += 64) {
    load_ab<K>(ap0, bp0, k0 + 32, a1, b1);
    mma_tile(acc, a0, b0);
    if (k0 + 64 < K) load_ab<K>(ap0, bp0, k0 + 64, a0, b0);
    mma_tile(acc, a1, b1);
  }

#pragma unroll
  for (int i = 0; i < 2; ++i)
#pragma unroll
    for (int j = 0; j < 4; ++j) {
      float* cp = C + (size_t)(m0 + i * 16 + 8 * hi) * N + n0 + j * 16 + col;
#pragma unroll
      for (int r = 0; r < 8; ++r) cp[(size_t)r * N] = acc[i][j][r];
    }
}

// ------------------------------------- RMS-norm + RoPE + scale + bf16 repack
// In : X (B,S,nh,128) f32.  Out: (B,nh,S,128) bf16.  One wave per row.
__global__ __launch_bounds__(256) void pack_qk(
    const float* __restrict__ X, const float* __restrict__ cs,
    const float* __restrict__ sn, const float* __restrict__ w,
    u16* __restrict__ out, int nh, float scale) {
  const int lane = threadIdx.x & 31;
  const int wid  = threadIdx.x >> 5;
  long rowid = (long)blockIdx.x * 8 + wid;
  int h = (int)(rowid % nh);
  long t = rowid / nh;
  int s = (int)(t % Sq);
  int b = (int)(t / Sq);

  const float4 x4 =
      *(const float4*)(X + (((size_t)(b * Sq + s)) * nh + h) * HD + lane * 4);
  float x[4] = {x4.x, x4.y, x4.z, x4.w};

  float ss = x[0] * x[0] + x[1] * x[1] + x[2] * x[2] + x[3] * x[3];
#pragma unroll
  for (int m = 1; m <= 16; m <<= 1) ss += __shfl_xor(ss, m);
  float rstd = rsqrtf(ss * (1.0f / HD) + 1e-6f);

  const float4 w4 = *(const float4*)(w + lane * 4);
  float xn[4] = {x[0] * rstd * w4.x, x[1] * rstd * w4.y,
                 x[2] * rstd * w4.z, x[3] * rstd * w4.w};

  if (s >= CTX) {
    int pos = s - CTX;
    const float4 c4 =
        *(const float4*)(cs + ((size_t)(b * MROT + pos)) * HD + lane * 4);
    const float4 s4 =
        *(const float4*)(sn + ((size_t)(b * MROT + pos)) * HD + lane * 4);
    float cv[4] = {c4.x, c4.y, c4.z, c4.w};
    float sv[4] = {s4.x, s4.y, s4.z, s4.w};
    float sgn = (lane < 16) ? -1.f : 1.f;  // d<64: -x[d+64], d>=64: +x[d-64]
#pragma unroll
    for (int i = 0; i < 4; ++i) {
      float oth = __shfl_xor(xn[i], 16);
      xn[i] = xn[i] * cv[i] + sgn * oth * sv[i];
    }
  }

  size_t dst = ((size_t)(b * nh + h) * Sq + s) * HD + lane * 4;
  ushort4 o;
  o.x = f2bf(xn[0] * scale);
  o.y = f2bf(xn[1] * scale);
  o.z = f2bf(xn[2] * scale);
  o.w = f2bf(xn[3] * scale);
  *(ushort4*)(out + dst) = o;
}

// ------------------------- V: (B,S,8,128) f32 -> transposed (B,8,128,S) bf16
__global__ void pack_v(const float* __restrict__ V, u16* __restrict__ Vt) {
  size_t gid = (size_t)blockIdx.x * blockDim.x + threadIdx.x;
  if (gid >= (size_t)Bc * Sq * NKV * HD) return;
  int d = (int)(gid & (HD - 1));
  size_t t = gid >> 7;
  int h = (int)(t & (NKV - 1));
  t >>= 3;
  int s = (int)(t & (Sq - 1));
  int b = (int)(t >> 11);
  Vt[((size_t)(b * NKV + h) * HD + d) * Sq + s] = f2bf(V[gid]);
}

// ------------------------------------------------------- flash attention ---
// One wave per 32 query rows (2 q-tiles) of one (b,h). K/V 32-key tiles are
// double-buffered in LDS and filled with GLOBAL_LOAD_ASYNC_TO_LDS_B128
// (ASYNCcnt): tile i+1 streams in while tile i is consumed via ds_load
// fragments. 32 async ops per tile -> s_wait_asynccnt 32 guarantees the
// current tile is resident while the next stays in flight. WAR on buffer
// reuse is covered by the DScnt waits the WMMAs already force.
__global__ __launch_bounds__(32) void flash_attn(
    const u16* __restrict__ Qb, const u16* __restrict__ Kb,
    const u16* __restrict__ Vt, const float* __restrict__ mask,
    u16* __restrict__ attb) {
  __shared__ u16 kbuf[2][32 * HD];  // K tile: 32 keys x 128d (row d-contig)
  __shared__ u16 vbuf[2][HD * 32];  // V tile: 128 d-rows x 32 keys
  __shared__ u16 pl[32 * 32];       // P relayout staging

  const int lane = threadIdx.x & 31;
  const int col  = lane & 15;
  const int hi   = lane >> 4;
  const int qt = blockIdx.x, h = blockIdx.y, b = blockIdx.z;
  const int kvh = h >> 1;  // GQA: 16 q-heads over 8 kv-heads

  const u16* kbase = Kb + (size_t)(b * NKV + kvh) * Sq * HD;
  const u16* vbase = Vt + (size_t)(b * NKV + kvh) * HD * Sq;

  // issue one 8KB K tile (16 async ops) + one 8KB V tile (16 async ops)
  auto issue_tile = [&](int n0, int buf) {
#pragma unroll
    for (int i = 0; i < 16; ++i) {           // K: contiguous 32*128 bf16
      int c = i * 32 + lane;                 // 512 x 16B chunks
      async_cp16(kbase + (size_t)n0 * HD + c * 8, &kbuf[buf][c * 8]);
    }
#pragma unroll
    for (int i = 0; i < 16; ++i) {           // V: 128 rows x 64B
      int c = i * 32 + lane;
      int d = c >> 2, part = c & 3;
      async_cp16(vbase + (size_t)d * Sq + n0 + part * 8,
                 &vbuf[buf][d * 32 + part * 8]);
    }
  };

  // Q fragments: 2 q-tiles x 4 d-steps, kept in registers the whole kernel.
  v16bf aq[2][4];
#pragma unroll
  for (int qi = 0; qi < 2; ++qi)
#pragma unroll
    for (int ks = 0; ks < 4; ++ks) {
      const u16* p = Qb +
          ((size_t)(b * NH + h) * Sq + qt * 32 + qi * 16 + col) * HD +
          ks * 32 + hi * 8;
      aq[qi][ks] = ld_frag(p, p + 16);
    }

  v8f acc[2][8];
  float m_i[2][8], l_i[2][8];
#pragma unroll
  for (int qi = 0; qi < 2; ++qi) {
#pragma unroll
    for (int dt = 0; dt < 8; ++dt)
      acc[qi][dt] = v8f{0.f, 0.f, 0.f, 0.f, 0.f, 0.f, 0.f, 0.f};
#pragma unroll
    for (int r = 0; r < 8; ++r) { m_i[qi][r] = -3.0e38f; l_i[qi][r] = 0.f; }
  }

  issue_tile(0, 0);  // prologue: tile 0 in flight

#pragma unroll 1
  for (int n0 = 0; n0 < Sq; n0 += 32) {
    const int cur = (n0 >> 5) & 1;
    const bool more = (n0 + 32) < Sq;
    if (more) issue_tile(n0 + 32, cur ^ 1);  // overlap next tile's DMA
    if (more) { WAIT_ASYNC(32); } else { WAIT_ASYNC(0); }

    // ---- scores: 2 q-tiles x 2 n-tiles; K fragments from LDS
    v8f s[2][2];
#pragma unroll
    for (int qi = 0; qi < 2; ++qi)
#pragma unroll
      for (int nt = 0; nt < 2; ++nt)
        s[qi][nt] = v8f{0.f, 0.f, 0.f, 0.f, 0.f, 0.f, 0.f, 0.f};
#pragma unroll
    for (int ks = 0; ks < 4; ++ks) {
      const u16* p0 = &kbuf[cur][(size_t)col * HD + ks * 32 + hi * 8];
      const u16* p1 = p0 + (size_t)16 * HD;
      v16bf bk0 = ld_frag(p0, p0 + 16);
      v16bf bk1 = ld_frag(p1, p1 + 16);
#pragma unroll
      for (int qi = 0; qi < 2; ++qi) {
        s[qi][0] = wmma_bf16(aq[qi][ks], bk0, s[qi][0]);
        s[qi][1] = wmma_bf16(aq[qi][ks], bk1, s[qi][1]);
      }
    }

    // ---- additive attention mask (B,1,S,S)
#pragma unroll
    for (int qi = 0; qi < 2; ++qi)
#pragma unroll
      for (int r = 0; r < 8; ++r) {
        int row = qt * 32 + qi * 16 + r + 8 * hi;
        const float* mrow = mask + ((size_t)b * Sq + row) * Sq + n0;
        s[qi][0][r] += mrow[col];
        s[qi][1][r] += mrow[16 + col];
      }

    // ---- online softmax (row stats across the 16 lanes holding each row)
#pragma unroll
    for (int qi = 0; qi < 2; ++qi) {
      float corr[8];
#pragma unroll
      for (int r = 0; r < 8; ++r) {
        float mx = fmaxf(s[qi][0][r], s[qi][1][r]);
        mx = fmaxf(mx, __shfl_xor(mx, 1));
        mx = fmaxf(mx, __shfl_xor(mx, 2));
        mx = fmaxf(mx, __shfl_xor(mx, 4));
        mx = fmaxf(mx, __shfl_xor(mx, 8));
        float mnew = fmaxf(m_i[qi][r], mx);
        float c  = __expf(m_i[qi][r] - mnew);
        float p0 = __expf(s[qi][0][r] - mnew);
        float p1 = __expf(s[qi][1][r] - mnew);
        float rs = p0 + p1;
        rs += __shfl_xor(rs, 1);
        rs += __shfl_xor(rs, 2);
        rs += __shfl_xor(rs, 4);
        rs += __shfl_xor(rs, 8);
        l_i[qi][r] = l_i[qi][r] * c + rs;
        m_i[qi][r] = mnew;
        corr[r] = c;
        s[qi][0][r] = p0;
        s[qi][1][r] = p1;
      }
#pragma unroll
      for (int dt = 0; dt < 8; ++dt)
#pragma unroll
        for (int r = 0; r < 8; ++r) acc[qi][dt][r] *= corr[r];
    }

    // ---- relayout P (C-layout) -> A-layout through LDS
#pragma unroll
    for (int qi = 0; qi < 2; ++qi)
#pragma unroll
      for (int r = 0; r < 8; ++r) {
        int mr = qi * 16 + r + 8 * hi;
        pl[mr * 32 + col]      = f2bf(s[qi][0][r]);
        pl[mr * 32 + 16 + col] = f2bf(s[qi][1][r]);
      }
    __syncthreads();
    v16bf ap[2];
#pragma unroll
    for (int qi = 0; qi < 2; ++qi) {
      const u16* pp = &pl[(qi * 16 + col) * 32 + hi * 8];
      ap[qi] = ld_frag(pp, pp + 16);
    }
    __syncthreads();

    // ---- out += P (16x32) * V (32x128); V fragments from LDS
#pragma unroll
    for (int dt = 0; dt < 8; ++dt) {
      const u16* pv = &vbuf[cur][(dt * 16 + col) * 32 + hi * 8];
      v16bf bv = ld_frag(pv, pv + 16);
#pragma unroll
      for (int qi = 0; qi < 2; ++qi)
        acc[qi][dt] = wmma_bf16(ap[qi], bv, acc[qi][dt]);
    }
  }

  // ---- normalize and emit bf16 at (b, s, h*128+d) for the Wo projection
#pragma unroll
  for (int qi = 0; qi < 2; ++qi)
#pragma unroll
    for (int r = 0; r < 8; ++r) {
      float inv = 1.0f / l_i[qi][r];
      int row = qt * 32 + qi * 16 + r + 8 * hi;
      size_t base = ((size_t)(b * Sq + row)) * (NH * HD) + h * HD;
#pragma unroll
      for (int dt = 0; dt < 8; ++dt)
        attb[base + dt * 16 + col] = f2bf(acc[qi][dt][r] * inv);
    }
}

// ---------------------------------------------------------------------------
extern "C" void kernel_launch(void* const* d_in, const int* in_sizes, int n_in,
                              void* d_out, int out_size, void* d_ws,
                              size_t ws_size, hipStream_t stream) {
  (void)in_sizes; (void)n_in; (void)out_size; (void)ws_size;
  const float* hidden = (const float*)d_in[0];
  // d_in[1] = target_hidden: only defines ctx=1024 in the reference; unused.
  const float* cosp = (const float*)d_in[2];
  const float* sinp = (const float*)d_in[3];
  const float* mask = (const float*)d_in[4];
  const float* Wq   = (const float*)d_in[5];
  const float* Wk   = (const float*)d_in[6];
  const float* Wv   = (const float*)d_in[7];
  const float* Wo   = (const float*)d_in[8];
  const float* qnw  = (const float*)d_in[9];
  const float* knw  = (const float*)d_in[10];
  float* out = (float*)d_out;
  char* ws = (char*)d_ws;

  const size_t MT = (size_t)Bc * Sq;  // 4096 token rows
  size_t off = 0;
  auto take = [&](size_t bytes) {
    size_t r = off;
    off += (bytes + 255) & ~(size_t)255;
    return r;
  };
  u16* Xb   = (u16*)(ws + take(MT * Hd * 2));
  u16* Wqb  = (u16*)(ws + take((size_t)NH * HD * Hd * 2));
  u16* Wkb  = (u16*)(ws + take((size_t)NKV * HD * Hd * 2));
  u16* Wvb  = (u16*)(ws + take((size_t)NKV * HD * Hd * 2));
  u16* Wob  = (u16*)(ws + take((size_t)Hd * NH * HD * 2));
  float* Qf = (float*)(ws + take(MT * NH * HD * 4));
  float* Kf = (float*)(ws + take(MT * NKV * HD * 4));
  float* Vf = (float*)(ws + take(MT * NKV * HD * 4));
  u16* Qb   = (u16*)(ws + take(MT * NH * HD * 2));
  u16* Kb   = (u16*)(ws + take(MT * NKV * HD * 2));
  u16* Vt   = (u16*)(ws + take(MT * NKV * HD * 2));
  u16* Attb = (u16*)Qf;  // Qf is dead after pack_qk; reuse for attention out

  // 1) fp32 -> bf16
  {
    size_t n;
    n = MT * Hd;               cvt_bf16<<<(n + 255) / 256, 256, 0, stream>>>(hidden, Xb, n);
    n = (size_t)NH * HD * Hd;  cvt_bf16<<<(n + 255) / 256, 256, 0, stream>>>(Wq, Wqb, n);
    n = (size_t)NKV * HD * Hd; cvt_bf16<<<(n + 255) / 256, 256, 0, stream>>>(Wk, Wkb, n);
    n = (size_t)NKV * HD * Hd; cvt_bf16<<<(n + 255) / 256, 256, 0, stream>>>(Wv, Wvb, n);
    n = (size_t)Hd * NH * HD;  cvt_bf16<<<(n + 255) / 256, 256, 0, stream>>>(Wo, Wob, n);
  }

  // 2) QKV projections (X * W^T); N,K are template constants
  gemm_bf16nt<NH * HD, Hd><<<dim3((NH * HD) / 256, MT / 64), 256, 0, stream>>>(
      Xb, Wqb, Qf);
  gemm_bf16nt<NKV * HD, Hd><<<dim3((NKV * HD) / 256, MT / 64), 256, 0, stream>>>(
      Xb, Wkb, Kf);
  gemm_bf16nt<NKV * HD, Hd><<<dim3((NKV * HD) / 256, MT / 64), 256, 0, stream>>>(
      Xb, Wvb, Vf);

  // 3) RMS-norm + RoPE + repack (scale folded into Q)
  pack_qk<<<(MT * NH) / 8, 256, 0, stream>>>(Qf, cosp, sinp, qnw, Qb, NH, SCALE);
  pack_qk<<<(MT * NKV) / 8, 256, 0, stream>>>(Kf, cosp, sinp, knw, Kb, NKV, 1.0f);
  pack_v<<<(MT * NKV * HD) / 256, 256, 0, stream>>>(Vf, Vt);

  // 4) flash attention (one wave per 32 query rows, async-LDS K/V pipeline)
  flash_attn<<<dim3(Sq / 32, NH, Bc), 32, 0, stream>>>(Qb, Kb, Vt, mask, Attb);

  // 5) output projection -> fp32 result
  gemm_bf16nt<Hd, NH * HD><<<dim3(Hd / 256, MT / 64), 256, 0, stream>>>(
      Attb, Wob, out);
}
